// Block_6322191860377
// MI455X (gfx1250) — matrix operand order repaired and checked by reading, not compiled, not run on hardware
//
#include <hip/hip_runtime.h>
#include <hip/hip_bf16.h>
#include <math.h>

// ---------------------------------------------------------------------------
// MI455X (gfx1250) fused kernel: 3x dilated 3x3 conv (18->6) + ReLU -> c[18],
// 1x1 convs (18->18 ReLU, 18->1), irregular-grid mean pool, broadcast,
// sigmoid, concat to 37 channels (+ branch2 as second output).
// Conv math on v_wmma_f32_16x16x32_bf16; A fragments gathered as aligned
// dword pairs (ds_load_b32), B fragments pre-packed once into workspace.
// ---------------------------------------------------------------------------

typedef __attribute__((ext_vector_type(16))) __bf16          v16bf;
typedef __attribute__((ext_vector_type(16))) unsigned short  v16u16;
typedef __attribute__((ext_vector_type(8)))  float           v8f;

union UA { v16u16 u; v16bf h; unsigned int w[8]; };

#define H_    1024
#define W_    1024
#define CIN   18
#define COUT  37
#define TROWS 4          // output rows per block
#define TCOLS 64         // output cols per block
#define HALO  3          // max dilation = 3
#define SROWS (TROWS + 2*HALO)   // 10
#define SCOLS (TCOLS + 2*HALO)   // 70
#define MAXC  16         // max pooling cells per dim (harness: 16x16)

#define NW3   (3*6*32*16)        // 9216 packed 3x3 B-fragment elements
#define NW1   (2*32*16)          // 1024 packed 1x1 B-fragment elements

// --- LDS carve (one 64B-aligned byte arena, phase-overlapped regions) ------
// phase1: sIn  ushort[10*70*18 + 8pad] = 25216 B @ 0
//         sW   ushort[9216]            = 18432 B @ 25216  (contig with sW1)
// phase2: cT   float [4*64*18]         = 18432 B @ 0      (overlays sIn)
//         accL float [4*16*19]         =  4864 B @ 25216  (overlays sW)
// both:   sW1  ushort[1024]            =  2048 B @ 43648
//         ints locR[17] locC[17] cidA[64]       @ 45696
//         cTb  ushort[4*64*18 + 8pad]  =  9232 B @ 46144  (bf16 copy of c)
#define OFF_SW    25216
#define OFF_SW1   43648
#define OFF_INT   45696
#define OFF_CTB   46144
#define SMEM_SZ   55424
#define ZPAD_U16  12600            // zeroed ushort slot in sIn region
#define ZPADB     4608             // zeroed ushort slot in cTb region

__device__ __forceinline__ unsigned short f2bf(float f) {
  unsigned u = __float_as_uint(f);                  // RNE fp32 -> bf16
  return (unsigned short)((u + 0x7FFFu + ((u >> 16) & 1u)) >> 16);
}

// searchsorted(loc, x, 'right') - 1, clamped to [0, ncell-1]
__device__ __forceinline__ int cell_of(const int* loc, int ncell, int x) {
  int id = 0;
  for (int i = 1; i <= ncell; ++i) id += (loc[i] <= x) ? 1 : 0;
  return (id < ncell) ? id : (ncell - 1);
}

// Patch offset for im2col K index (fully constant-folds in unrolled loops).
__device__ __forceinline__ constexpr int eoff(int K, int dil) {
  int tap = K / CIN, ci = K % CIN;
  int ky = tap / 3, kx = tap % 3;
  return (((ky - 1)*dil + HALO)*SCOLS + ((kx - 1)*dil + HALO))*CIN + ci;
}

// ---- one-shot: pack 3x3 + 1x1 weights into WMMA B-fragment images --------
__global__ void prep_weights_kernel(const float* __restrict__ w1,
                                    const float* __restrict__ w2,
                                    const float* __restrict__ w3,
                                    const float* __restrict__ wb1,
                                    const float* __restrict__ wb2,
                                    unsigned short* __restrict__ wpack) {
  int i = blockIdx.x*blockDim.x + threadIdx.x;
  if (i < NW3) {                       // 3x3 convs: K=192 pad, N=16 pad
    int e = i & 15, l = (i >> 4) & 31, s = (i >> 9) % 6, d = i / 3072;
    int n = l & 15, gg = l >> 4;
    int K = 32*s + ((e < 8) ? (8*gg + e) : (16 + 8*gg + (e - 8)));  // B layout: lane=(Kgroup,N), table K order
    float v = 0.f;
    if (K < 162 && n < 6) {
      int tap = K / CIN, ci = K - tap*CIN;
      const float* wp = (d == 0) ? w1 : ((d == 1) ? w2 : w3);
      v = wp[(tap*CIN + ci)*6 + n];
    }
    wpack[i] = f2bf(v);
  } else if (i < NW3 + NW1) {          // 1x1: Ntile0=wb1[:,0:16]; Ntile1 rest
    int j = i - NW3;
    int e = j & 15, l = (j >> 4) & 31, t = j >> 9;
    int n = l & 15, gg = l >> 4;
    int K = (e < 8) ? (8*gg + e) : (16 + 8*gg + (e - 8));
    float v = 0.f;
    if (K < CIN) {
      if (t == 0)      v = wb1[K*CIN + n];
      else if (n < 2)  v = wb1[K*CIN + 16 + n];
      else if (n == 2) v = wb2[K];
    }
    wpack[i] = f2bf(v);
  }
}

__global__ __launch_bounds__(256)
void fused_conv_kernel(const float* __restrict__ in,
                       const int*   __restrict__ rloc,
                       const int*   __restrict__ cloc,
                       const unsigned short* __restrict__ wpack,
                       const float* __restrict__ b1, const float* __restrict__ b2,
                       const float* __restrict__ b3,
                       const float* __restrict__ bb1, const float* __restrict__ bb2,
                       float* __restrict__ out, float* __restrict__ wsum,
                       int mc, int nc) {
  __shared__ __align__(64) unsigned char smem[SMEM_SZ];
  unsigned short* sIn  = (unsigned short*)smem;               // phase1
  float*          cT   = (float*)smem;                        // phase2
  unsigned short* sW   = (unsigned short*)(smem + OFF_SW);    // phase1
  float*          accL = (float*)(smem + OFF_SW);             // phase2
  unsigned short* sW1  = (unsigned short*)(smem + OFF_SW1);
  int*            locR = (int*)(smem + OFF_INT);
  int*            locC = locR + (MAXC + 1);
  int*            cidA = locC + (MAXC + 1);
  unsigned short* cTb  = (unsigned short*)(smem + OFF_CTB);

  const int tid  = threadIdx.x;
  const int lane = tid & 31;
  const int wv   = tid >> 5;          // 8 waves
  const int b    = blockIdx.z;
  const int h0   = blockIdx.y * TROWS;
  const int w0   = blockIdx.x * TCOLS;
  const int m    = lane & 15;         // M row within WMMA tile
  const int g    = (lane >> 4) & 1;   // lane group (K half selector)

  // Each wave owns one output row and two adjacent 16-px M-tiles:
  const int row    = wv >> 1;
  const int mtbase = (wv & 1) * 2;

  // ---- stage grid-cell boundaries ----------------------------------------
  if (tid <= mc)                       locR[tid]      = rloc[tid];
  if (tid >= 64 && tid - 64 <= nc)     locC[tid - 64] = cloc[tid - 64];

  // ---- stage pre-packed B fragments: one linear 20480B uint4 copy --------
  {
    const uint4* src = (const uint4*)wpack;
    uint4* dst = (uint4*)(smem + OFF_SW);      // sW then sW1, contiguous
    for (int i = tid; i < (NW3 + NW1)*2/16; i += 256) dst[i] = src[i];
  }
  // ---- stage input tile (halo + zero pad) as packed bf16 pairs -----------
  for (int i2 = tid; i2 < SROWS*SCOLS*CIN/2; i2 += 256) {
    int i = i2*2;
    int ci = i % CIN; int rest = i / CIN;
    int col = rest % SCOLS; int rw = rest / SCOLS;
    int gh = h0 + rw - HALO, gw = w0 + col - HALO;
    unsigned int pack = 0;
    if (gh >= 0 && gh < H_ && gw >= 0 && gw < W_) {
      const float2 v = *(const float2*)&in[(((size_t)b*H_ + gh)*W_ + gw)*CIN + ci];
      pack = (unsigned)f2bf(v.x) | ((unsigned)f2bf(v.y) << 16);
    }
    *(unsigned int*)(sIn + i) = pack;
  }
  if (tid < 8) sIn[ZPAD_U16 + tid] = 0;          // zero slot for OOB-K gathers
  // speculative L2 prefetch of next row-tile of the input
  if (blockIdx.y + 1 < gridDim.y && tid < 64) {
    const float* nxt = in + (((size_t)b*H_ + (h0 + TROWS))*W_ + w0)*CIN;
    __builtin_prefetch(nxt + tid*32, 0, 1);
  }
  __syncthreads();

  // column -> pool-cell lookup (consumed after next barrier)
  if (tid < TCOLS) cidA[tid] = cell_of(locC, nc, w0 + tid);

  // ---- phase 1: implicit-GEMM dilated convs via WMMA ---------------------
  // K = 162 -> 6 bf16 k-steps of 32; both M-tiles share every B fragment and
  // issue back-to-back independent WMMAs (hides the bf16 RAW hazard).
  v8f zero8 = {0.f,0.f,0.f,0.f,0.f,0.f,0.f,0.f};
  v8f acc[2][3];
#pragma unroll
  for (int t = 0; t < 2; ++t)
#pragma unroll
    for (int d = 0; d < 3; ++d) acc[t][d] = zero8;

  const int base = (row*SCOLS + mtbase*16 + m) * CIN;   // pixel base (even)
#pragma unroll
  for (int d = 0; d < 3; ++d) {
    const int dil = d + 1;
#pragma unroll
    for (int s = 0; s < 6; ++s) {
      union UA B, A0, A1;
      B.u = *(const v16u16*)(sW + ((d*6 + s)*32 + lane)*16);
#pragma unroll
      for (int p = 0; p < 8; ++p) {          // dword pairs: elems (2p, 2p+1)
        const int K0 = 32*s + ((2*p < 8) ? 2*p : (8 + 2*p));  // even
        if (K0 + 9 < 162) {                  // whole pair valid, both groups
          const int c0 = eoff(K0, dil), c1 = eoff(K0 + 8, dil);
          const int a = base + (g ? c1 : c0);          // even -> 4B aligned
          A0.w[p] = *(const unsigned int*)(sIn + a);
          A1.w[p] = *(const unsigned int*)(sIn + a + 16*CIN);
        } else if (K0 < 162) {               // s==5, p==0: group 0 only
          const int c0 = eoff(K0, dil);
          A0.w[p] = *(const unsigned int*)(sIn + (g ? ZPAD_U16 : (base + c0)));
          A1.w[p] = *(const unsigned int*)(sIn + (g ? ZPAD_U16 : (base + c0 + 16*CIN)));
        } else {                             // compile-time zero
          A0.w[p] = 0; A1.w[p] = 0;
        }
      }
      acc[0][d] = __builtin_amdgcn_wmma_f32_16x16x32_bf16(
          false, A0.h, false, B.h, (short)0, acc[0][d], false, false);
      acc[1][d] = __builtin_amdgcn_wmma_f32_16x16x32_bf16(
          false, A1.h, false, B.h, (short)0, acc[1][d], false, false);
    }
  }
  __syncthreads();   // sIn / sW dead beyond this point

  // ---- write c tile (bias + ReLU) to LDS in fp32 + bf16, zero pool accs --
#pragma unroll
  for (int t = 0; t < 2; ++t) {
    int mt = mtbase + t;
    int n = lane & 15;
    if (n < 6) {
      float bi0 = b1[n], bi1 = b2[n], bi2 = b3[n];
#pragma unroll
      for (int r = 0; r < 8; ++r) {
        int px = mt*16 + r + 8*g;                 // D layout: M = r + 8*g
        int idx = (row*TCOLS + px)*CIN;
        float v0 = acc[t][0][r] + bi0; v0 = v0 > 0.f ? v0 : 0.f;
        float v1 = acc[t][1][r] + bi1; v1 = v1 > 0.f ? v1 : 0.f;
        float v2 = acc[t][2][r] + bi2; v2 = v2 > 0.f ? v2 : 0.f;
        cT[idx + n]       = v0;  cTb[idx + n]      = f2bf(v0);
        cT[idx + 6 + n]   = v1;  cTb[idx + 6 + n]  = f2bf(v1);
        cT[idx + 12 + n]  = v2;  cTb[idx + 12 + n] = f2bf(v2);
      }
    }
  }
  if (tid < 8) cTb[ZPADB + tid] = 0;              // zero slot for 1x1 gather
  for (int i = tid; i < TROWS*MAXC*19; i += 256) accL[i] = 0.f;
  __syncthreads();

  // ---- phase 2a: coalesced store of c into output channels [18..36) -----
  for (int i = tid; i < TROWS*TCOLS*CIN; i += 256) {
    int ch = i % CIN; int rest = i / CIN;
    int px = rest % TCOLS; int rw = rest / TCOLS;
    size_t pix = ((size_t)b*H_ + (h0 + rw))*W_ + (w0 + px);
    out[pix*COUT + CIN + ch] = cT[i];
  }

  // ---- phase 2b: 1x1 convs via WMMA + pooled-sum accumulation ------------
  const int rbase = cell_of(locR, mc, h0);
  const int rslot = cell_of(locR, mc, h0 + row) - rbase;  // row cell (0..3)
  union UA B0, B1;
  B0.u = *(const v16u16*)(sW1 + lane*16);
  B1.u = *(const v16u16*)(sW1 + (32 + lane)*16);
  const int base2 = (row*TCOLS + mtbase*16 + m)*CIN;      // even
#pragma unroll
  for (int t = 0; t < 2; ++t) {
    union UA A;
#pragma unroll
    for (int p = 0; p < 8; ++p) {
      if (p < 4) {                                // K = 2p(+1) + 8g, valid
        A.w[p] = *(const unsigned int*)(cTb + base2 + t*16*CIN + 2*p + 8*g);
      } else if (p == 4) {                        // K = 16,17 (g1 invalid)
        int a = g ? ZPADB : (base2 + t*16*CIN + 16);
        A.w[p] = *(const unsigned int*)(cTb + a);
      } else {                                    // K >= 18: zero
        A.w[p] = 0;
      }
    }
    v8f p0 = __builtin_amdgcn_wmma_f32_16x16x32_bf16(
        false, A.h, false, B0.h, (short)0, zero8, false, false);
    v8f p1 = __builtin_amdgcn_wmma_f32_16x16x32_bf16(
        false, A.h, false, B1.h, (short)0, zero8, false, false);

    int mt = mtbase + t;
    int n = lane & 15;
    float biasA = bb1[n];
#pragma unroll
    for (int r = 0; r < 8; ++r) {
      int px  = mt*16 + r + 8*g;
      int cid = cidA[px];
      float va = p0[r] + biasA; va = va > 0.f ? va : 0.f;    // branch1 ch n
      atomicAdd(&accL[(rslot*MAXC + cid)*19 + n], va);
      if (n < 2) {                                           // branch1 ch 16+n
        float vb = p1[r] + bb1[16 + n]; vb = vb > 0.f ? vb : 0.f;
        atomicAdd(&accL[(rslot*MAXC + cid)*19 + 16 + n], vb);
      } else if (n == 2) {                                   // branch2 (pre-sigmoid)
        float vb = p1[r] + bb2[0];
        atomicAdd(&accL[(rslot*MAXC + cid)*19 + 18], vb);
      }
    }
  }
  __syncthreads();

  // ---- flush per-block cell sums to global accumulators ------------------
  for (int i = tid; i < TROWS*MAXC*19; i += 256) {
    float v = accL[i];
    if (v != 0.f) {
      int ch = i % 19; int rest = i / 19;
      int cid = rest % MAXC; int rs = rest / MAXC;
      int rid = rbase + rs;
      if (rid < mc && cid < nc)
        atomicAdd(&wsum[(((size_t)b*mc + rid)*nc + cid)*19 + ch], v);
    }
  }
}

__global__ void zero_kernel(float* __restrict__ p, int n) {
  int i = blockIdx.x*blockDim.x + threadIdx.x;
  if (i < n) p[i] = 0.f;
}

__global__ void pool_finalize_kernel(const int* __restrict__ rloc,
                                     const int* __restrict__ cloc,
                                     float* __restrict__ wsum,
                                     int B, int mc, int nc) {
  int i = blockIdx.x*blockDim.x + threadIdx.x;
  if (i >= B*mc*nc) return;
  int ci = i % nc; int rest = i / nc;
  int ri = rest % mc;
  int cnt_r = (ri == mc - 1) ? (H_ - rloc[ri]) : (rloc[ri + 1] - rloc[ri]);
  int cnt_c = (ci == nc - 1) ? (W_ - cloc[ci]) : (cloc[ci + 1] - cloc[ci]);
  int cnt = cnt_r * cnt_c; if (cnt < 1) cnt = 1;
  float inv = 1.f / (float)cnt;
  float* p = &wsum[(size_t)i * 19];
  for (int ch = 0; ch < 19; ++ch) {
    float mval = p[ch] * inv;
    if (ch == 18) mval = 1.f / (1.f + __expf(-mval));   // sigmoid(branch2 mean)
    p[ch] = mval;
  }
}

__global__ __launch_bounds__(256)
void broadcast_kernel(const int* __restrict__ rloc, const int* __restrict__ cloc,
                      const float* __restrict__ pool,
                      float* __restrict__ out, float* __restrict__ out2,
                      int B, int mc, int nc) {
  __shared__ int lr[MAXC + 1], lc[MAXC + 1];
  if (threadIdx.x <= mc)                           lr[threadIdx.x]      = rloc[threadIdx.x];
  if (threadIdx.x >= 32 && threadIdx.x - 32 <= nc) lc[threadIdx.x - 32] = cloc[threadIdx.x - 32];
  __syncthreads();
  size_t total = (size_t)B * H_ * W_ * 19;
  for (size_t e = (size_t)blockIdx.x*blockDim.x + threadIdx.x; e < total;
       e += (size_t)gridDim.x * blockDim.x) {
    int k = (int)(e % 19);
    size_t pix = e / 19;
    int w = (int)(pix % W_);
    size_t rest = pix / W_;
    int h = (int)(rest % H_);
    int b = (int)(rest / H_);
    int rid = cell_of(lr, mc, h);
    int cid = cell_of(lc, nc, w);
    float pv = pool[(((size_t)b*mc + rid)*nc + cid)*19 + k];
    if (k < 18) out[pix*COUT + k] = pv;                 // branch1 broadcast
    else { out[pix*COUT + 36] = pv; out2[pix] = pv; }   // sigmoid(branch2)
  }
}

extern "C" void kernel_launch(void* const* d_in, const int* in_sizes, int n_in,
                              void* d_out, int out_size, void* d_ws, size_t ws_size,
                              hipStream_t stream) {
  const float* in  = (const float*)d_in[0];
  const int* rloc  = (const int*)d_in[1];
  const int* cloc  = (const int*)d_in[2];
  const float* w1  = (const float*)d_in[3];  const float* b1  = (const float*)d_in[4];
  const float* w2  = (const float*)d_in[5];  const float* b2  = (const float*)d_in[6];
  const float* w3  = (const float*)d_in[7];  const float* b3  = (const float*)d_in[8];
  const float* wb1 = (const float*)d_in[9];  const float* bb1 = (const float*)d_in[10];
  const float* wb2 = (const float*)d_in[11]; const float* bb2 = (const float*)d_in[12];

  int mc = in_sizes[1] - 1;                  // pooling cells (rows)
  int nc = in_sizes[2] - 1;                  // pooling cells (cols)
  int B  = in_sizes[0] / (H_ * W_ * CIN);

  float* out  = (float*)d_out;
  float* out2 = out + (size_t)B * H_ * W_ * COUT;   // second tuple output
  float* wsum = (float*)d_ws;                       // [B][mc][nc][19] sums->means
  int nws = B * mc * nc * 19;
  size_t wpackOff = (((size_t)nws * 4) + 63) & ~(size_t)63;
  unsigned short* wpack = (unsigned short*)((char*)d_ws + wpackOff);

  prep_weights_kernel<<<(NW3 + NW1 + 255)/256, 256, 0, stream>>>(
      w1, w2, w3, wb1, wb2, wpack);
  zero_kernel<<<(nws + 255)/256, 256, 0, stream>>>(wsum, nws);

  dim3 grid(W_/TCOLS, H_/TROWS, B);
  fused_conv_kernel<<<grid, 256, 0, stream>>>(in, rloc, cloc, wpack,
      b1, b2, b3, bb1, bb2, out, wsum, mc, nc);

  pool_finalize_kernel<<<(B*mc*nc + 63)/64, 64, 0, stream>>>(rloc, cloc, wsum, B, mc, nc);

  size_t totalC = (size_t)B * H_ * W_ * 19;
  int blocksC = (int)((totalC + 255) / 256);
  broadcast_kernel<<<blocksC, 256, 0, stream>>>(rloc, cloc, wsum, out, out2, B, mc, nc);
}
